// TransConv_10694468567476
// MI455X (gfx1250) — matrix operand order repaired and checked by reading, not compiled, not run on hardware
//
#include <hip/hip_runtime.h>
#include <hip/hip_fp16.h>

typedef _Float16 v16h __attribute__((ext_vector_type(16)));
typedef _Float16 v8h  __attribute__((ext_vector_type(8)));
typedef float    v8f  __attribute__((ext_vector_type(8)));

#define IN_F   128
#define OUT_F  64
#define HEADS  8
#define DK     8
#define QKV_C  192   // q|k|v concatenated output columns

// ---------- helpers: order-preserving float<->uint map for atomicMax ----------
__device__ __forceinline__ unsigned mapf(float f) {
  unsigned b = __float_as_uint(f);
  return (b & 0x80000000u) ? ~b : (b | 0x80000000u);
}
__device__ __forceinline__ float unmapf(unsigned u) {
  unsigned b = (u & 0x80000000u) ? (u & 0x7FFFFFFFu) : ~u;
  return __uint_as_float(b);
}

// ---------- zero outputs + segment buffers (deterministic re-init) ----------
__global__ void init_zero(float* __restrict__ out, unsigned* __restrict__ segmax,
                          float* __restrict__ segsum, int n_out, int n_seg) {
  int i = blockIdx.x * blockDim.x + threadIdx.x;
  int stride = gridDim.x * blockDim.x;
  for (int j = i; j < n_out; j += stride) out[j] = 0.f;
  for (int j = i; j < n_seg; j += stride) { segmax[j] = 0u; segsum[j] = 0.f; }
}

// ---------- fuse Wq|Wk|Wv -> f16 [192][128], gather biases ----------
__global__ void prep_weights(const float* __restrict__ Wq, const float* __restrict__ bq,
                             const float* __restrict__ Wk, const float* __restrict__ bk,
                             const float* __restrict__ Wv, const float* __restrict__ bv,
                             _Float16* __restrict__ w16, float* __restrict__ bias) {
  int idx = blockIdx.x * blockDim.x + threadIdx.x;
  if (idx >= QKV_C * IN_F) return;
  int col = idx >> 7;          // output column 0..191
  int kk  = idx & (IN_F - 1);  // input feature 0..127
  float w, b;
  if (col < 64)       { w = Wq[col * IN_F + kk];         b = bq[col]; }
  else if (col < 128) { w = Wk[(col - 64) * IN_F + kk];  b = bk[col - 64]; }
  else                { w = Wv[(col - 128) * IN_F + kk]; b = bv[col - 128]; }
  w16[idx] = (_Float16)w;
  if (kk == 0) bias[col] = b;
}

// ---------- fused QKV GEMM via v_wmma_f32_16x16x32_f16 ----------
// grid.x = ceil(N/16) row tiles, grid.y = 3; 4 waves/block -> 12 column tiles of 16
__global__ __launch_bounds__(128)
void qkv_gemm(const float* __restrict__ x, const _Float16* __restrict__ w16,
              const float* __restrict__ bias, float* __restrict__ qkv, int nrows) {
  __shared__ _Float16 xa[16][136];   // padded rows (272B) -> conflict-free ds_load_b128

  const int row0 = blockIdx.x * 16;

  // stage + convert the 16x128 A tile to f16 in LDS (coalesced)
  #pragma unroll
  for (int j = 0; j < 16; ++j) {
    int idx = j * 128 + threadIdx.x;
    int r = idx >> 7, c = idx & (IN_F - 1);
    int gr = row0 + r;
    if (gr >= nrows) gr = nrows - 1;   // clamp (N is a multiple of 16 anyway)
    xa[r][c] = (_Float16)x[(size_t)gr * IN_F + c];
  }
  __syncthreads();

  const int wave   = threadIdx.x >> 5;
  const int lane   = threadIdx.x & 31;
  const int tile_n = blockIdx.y * 4 + wave;       // 0..11
  const int r      = lane & 15;                   // A row
  const int kh     = lane >> 4;                   // lane half selects K sub-range
  const int gcol   = tile_n * 16 + (lane & 15);   // output column 0..191

  v8f acc = {};
  #pragma unroll
  for (int step = 0; step < 4; ++step) {          // K = 128 in 4 chunks of 32
    // A operand: lanes 0-15 K=[0..15], lanes 16-31 K=[16..31] (matches B)
    int ka = step * 32 + kh * 16;
    v8h lo = *(const v8h*)&xa[r][ka];
    v8h hi = *(const v8h*)&xa[r][ka + 8];
    v16h a;
    #pragma unroll
    for (int i = 0; i < 8; ++i) { a[i] = lo[i]; a[8 + i] = hi[i]; }
    // B operand: column gcol, lanes 0-15 K=[0..15], lanes 16-31 K=[16..31]
    int kb = step * 32 + kh * 16;
    v16h b = *(const v16h*)&w16[(size_t)gcol * IN_F + kb];
    acc = __builtin_amdgcn_wmma_f32_16x16x32_f16(
        /*neg_a=*/false, a, /*neg_b=*/false, b,
        /*c_mod=*/(short)0, acc, /*reuse_a=*/false, /*reuse_b=*/false);
  }

  // D layout: VGPR j -> M = j + 8*(lane>=16), N = lane&15
  const float bb = bias[gcol];
  float* op = qkv + (size_t)(row0 + kh * 8) * QKV_C + gcol;
  if (row0 + 16 <= nrows) {
    // full tile: straight-line coalesced stores, no per-row EXEC churn
    #pragma unroll
    for (int j = 0; j < 8; ++j) op[(size_t)j * QKV_C] = acc[j] + bb;
  } else {
    const int mbase = kh * 8;
    for (int j = 0; j < 8; ++j)
      if (row0 + mbase + j < nrows) op[(size_t)j * QKV_C] = acc[j] + bb;
  }
}

// ---------- per-(edge,head) scores + segment max ----------
__global__ void scores_kernel(const int* __restrict__ edge, const float* __restrict__ qkv,
                              float* __restrict__ sc, unsigned* __restrict__ segmax, int E_) {
  int t = blockIdx.x * blockDim.x + threadIdx.x;
  if (t >= E_ * HEADS) return;
  int e = t >> 3, h = t & 7;
  int src = edge[e];
  int dst = edge[E_ + e];
  const float* qp = qkv + (size_t)src * QKV_C + h * DK;          // q block
  const float* kp = qkv + (size_t)dst * QKV_C + OUT_F + h * DK;  // k block
  float s = 0.f;
  #pragma unroll
  for (int d = 0; d < DK; ++d) s += qp[d] * kp[d];
  s *= 0.35355339059327373f;  // 1/sqrt(8)
  sc[t] = s;
  atomicMax(&segmax[src * HEADS + h], mapf(s));
}

// ---------- exp(score - max) + segment sum ----------
__global__ void expsum_kernel(const int* __restrict__ edge, float* __restrict__ sc,
                              const unsigned* __restrict__ segmax,
                              float* __restrict__ segsum, int E_) {
  int t = blockIdx.x * blockDim.x + threadIdx.x;
  if (t >= E_ * HEADS) return;
  int e = t >> 3, h = t & 7;
  int src = edge[e];
  float m = unmapf(segmax[src * HEADS + h]);
  float ex = __expf(sc[t] - m);
  sc[t] = ex;
  atomicAdd(&segsum[src * HEADS + h], ex);
}

// ---------- normalize + scatter-add v messages ----------
__global__ void aggregate_kernel(const int* __restrict__ edge, const float* __restrict__ qkv,
                                 const float* __restrict__ sc, const float* __restrict__ segsum,
                                 float* __restrict__ out, int E_) {
  int t = blockIdx.x * blockDim.x + threadIdx.x;
  if (t >= E_ * HEADS) return;
  int e = t >> 3, h = t & 7;
  int src = edge[e];
  int dst = edge[E_ + e];
  float w = sc[t] / (segsum[src * HEADS + h] + 1e-16f);
  const float* vp = qkv + (size_t)dst * QKV_C + 2 * OUT_F + h * DK;  // v block
  __builtin_prefetch(vp, 0, 0);                                      // global_prefetch_b8
  float* op = out + (size_t)src * OUT_F + h * DK;
  #pragma unroll
  for (int d = 0; d < DK; ++d) atomicAdd(&op[d], w * vp[d]);
}

extern "C" void kernel_launch(void* const* d_in, const int* in_sizes, int n_in,
                              void* d_out, int out_size, void* d_ws, size_t ws_size,
                              hipStream_t stream) {
  const float* x  = (const float*)d_in[0];
  const int* edge = (const int*)d_in[1];
  const float* Wq = (const float*)d_in[2];
  const float* bq = (const float*)d_in[3];
  const float* Wk = (const float*)d_in[4];
  const float* bk = (const float*)d_in[5];
  const float* Wv = (const float*)d_in[6];
  const float* bv = (const float*)d_in[7];
  float* out = (float*)d_out;

  const int n = in_sizes[0] / IN_F;  // 100000 nodes
  const int e = in_sizes[1] / 2;     // 1600000 edges

  // workspace carve-up
  char* ws = (char*)d_ws;
  size_t off = 0;
  _Float16* w16 = (_Float16*)(ws + off); off += (size_t)QKV_C * IN_F * sizeof(_Float16);
  off = (off + 255) & ~(size_t)255;
  float* bias   = (float*)(ws + off);    off += (size_t)QKV_C * sizeof(float);
  off = (off + 255) & ~(size_t)255;
  float* qkv    = (float*)(ws + off);    off += (size_t)n * QKV_C * sizeof(float);
  off = (off + 255) & ~(size_t)255;
  float* sc     = (float*)(ws + off);    off += (size_t)e * HEADS * sizeof(float);
  off = (off + 255) & ~(size_t)255;
  unsigned* segmax = (unsigned*)(ws + off); off += (size_t)n * HEADS * sizeof(unsigned);
  off = (off + 255) & ~(size_t)255;
  float* segsum = (float*)(ws + off);

  init_zero<<<1024, 256, 0, stream>>>(out, segmax, segsum, n * OUT_F, n * HEADS);

  prep_weights<<<(QKV_C * IN_F + 255) / 256, 256, 0, stream>>>(Wq, bq, Wk, bk, Wv, bv, w16, bias);

  dim3 g1((n + 15) / 16, 3);
  qkv_gemm<<<g1, 128, 0, stream>>>(x, w16, bias, qkv, n);

  const int eh = e * HEADS;
  const int blocks = (eh + 255) / 256;
  scores_kernel<<<blocks, 256, 0, stream>>>(edge, qkv, sc, segmax, e);
  expsum_kernel<<<blocks, 256, 0, stream>>>(edge, sc, segmax, segsum, e);
  aggregate_kernel<<<blocks, 256, 0, stream>>>(edge, qkv, sc, segsum, out, e);
}
